// MultiHeadAttention_53738630808203
// MI455X (gfx1250) — matrix operand order repaired and checked by reading, not compiled, not run on hardware
//
#include <hip/hip_runtime.h>
#include <hip/hip_bf16.h>
#include <math.h>
#include <stdint.h>

typedef _Float16 f16;
typedef __attribute__((ext_vector_type(16))) _Float16 v16h;
typedef __attribute__((ext_vector_type(8)))  _Float16 v8h;
typedef __attribute__((ext_vector_type(8)))  float    v8f;

#define D_MODEL 512
#define NUM_HEADS 8
#define HEAD_DIM 64
#define SEQ 4096
#define BATCH 2
#define MROWS (BATCH * SEQ)   // 8192

// ---------------------------------------------------------------------------
// Fragment loaders (CDNA5 wave32 WMMA layouts, ISA 7.12.2)
// ---------------------------------------------------------------------------
__device__ __forceinline__ v16h load16(const f16* __restrict__ p) {
  const v8h* q = (const v8h*)p;
  v8h lo = q[0], hi = q[1];
  return __builtin_shufflevector(lo, hi, 0,1,2,3,4,5,6,7,8,9,10,11,12,13,14,15);
}
__device__ __forceinline__ v16h load8x2(const f16* __restrict__ p) {
  const v8h* q = (const v8h*)p;
  v8h lo = q[0], hi = q[2];
  return __builtin_shufflevector(lo, hi, 0,1,2,3,4,5,6,7,8,9,10,11,12,13,14,15);
}
__device__ __forceinline__ v16h load_a_any(const float* __restrict__ A, long ld,
                                           int mb, int kb) {
  const int l16 = threadIdx.x & 15, hlf = (threadIdx.x >> 4) & 1;
  const float* row = A + (long)(mb + l16) * ld + kb + hlf * 8;
  v16h a;
#pragma unroll
  for (int i = 0; i < 8; ++i) a[i] = (f16)row[i];
#pragma unroll
  for (int i = 0; i < 8; ++i) a[8 + i] = (f16)row[16 + i];
  return a;
}
__device__ __forceinline__ v16h load_a_any(const f16* __restrict__ A, long ld,
                                           int mb, int kb) {
  const int l16 = threadIdx.x & 15, hlf = (threadIdx.x >> 4) & 1;
  return load8x2(A + (long)(mb + l16) * ld + kb + hlf * 8);
}

// CDNA5 async global->LDS copy, 16 bytes per lane (ASYNCcnt-tracked)
__device__ __forceinline__ void async_copy16(const f16* __restrict__ g, f16* l) {
  unsigned lds = (unsigned)(uintptr_t)l;   // LDS aperture: addr[31:0] = LDS offset
  asm volatile("global_load_async_to_lds_b128 %0, %1, off"
               :: "v"(lds), "v"(g) : "memory");
}
__device__ __forceinline__ void wait_async0() {
  asm volatile("s_wait_asynccnt 0" ::: "memory");
}

__global__ void cvt_f32_f16(const float* __restrict__ in, f16* __restrict__ out, int n) {
  int i = blockIdx.x * blockDim.x + threadIdx.x;
  if (i < n) out[i] = (f16)in[i];
}

// ---------------------------------------------------------------------------
// GEMM: C[M,N] = A[M,K] * W[N,K]^T + bias.  One wave = 16x64 tile.
// OUTMODE 0: f16 row-major   1: f16 V-transposed [B,H,64,S]   2: f32 row-major
// ---------------------------------------------------------------------------
template <typename AT, int OUTMODE>
__global__ void __launch_bounds__(256)
gemm_wmma_kernel(const AT* __restrict__ A, const f16* __restrict__ W,
                 const float* __restrict__ bias, void* __restrict__ out,
                 int M, int N, int K, float scale) {
  const int wave = threadIdx.x >> 5;
  const int n64 = N >> 6;
  const int tile = blockIdx.x * (blockDim.x >> 5) + wave;   // wave-uniform
  if (tile >= (M >> 4) * n64) return;
  const int mbase = (tile / n64) << 4;
  const int n0 = (tile % n64) << 6;
  const int l16 = threadIdx.x & 15, hlf = (threadIdx.x >> 4) & 1;

  const f16* wbase = W + (long)(n0 + l16) * K + hlf * 8;
  v8f acc[4];
#pragma unroll
  for (int c = 0; c < 4; ++c) { v8f z = {}; acc[c] = z; }

  for (int kb = 0; kb < K; kb += 32) {
    if (kb + 32 < K) __builtin_prefetch(wbase + kb + 32, 0, 1); // global_prefetch_b8
    const v16h a = load_a_any(A, (long)K, mbase, kb);
#pragma unroll
    for (int c = 0; c < 4; ++c) {
      const v16h b = load8x2(wbase + (long)(c * 16) * K + kb);
      acc[c] = __builtin_amdgcn_wmma_f32_16x16x32_f16(false, a, false, b,
                                                      (short)0, acc[c], false, false);
    }
  }

#pragma unroll
  for (int c = 0; c < 4; ++c) {
    const int col = n0 + c * 16 + l16;
    const float bn = bias[col];
#pragma unroll
    for (int r = 0; r < 8; ++r) {
      const int row = mbase + r + 8 * hlf;
      const float vv = (acc[c][r] + bn) * scale;
      if constexpr (OUTMODE == 0) {
        ((f16*)out)[(long)row * N + col] = (f16)vv;
      } else if constexpr (OUTMODE == 2) {
        ((float*)out)[(long)row * N + col] = vv;
      } else {
        const int bb = row >> 12, s = row & (SEQ - 1);
        const int h = col >> 6, d = col & 63;
        ((f16*)out)[((long)(bb * NUM_HEADS + h) * HEAD_DIM + d) * SEQ + s] = (f16)vv;
      }
    }
  }
}

// ---------------------------------------------------------------------------
// Flash attention. Workgroup = 4 waves = 64 q-rows of one (b,h).
// KV blocks of 64 staged in double-buffered LDS via async global->LDS copies.
// ---------------------------------------------------------------------------
template <int BIAS_MODE>
__device__ __forceinline__ void attn_block64(
    int kv0, int qbase, const float* __restrict__ rp, float bc,
    const f16 (*__restrict__ Kt)[64], const f16 (*__restrict__ Vs)[64],
    f16 (*__restrict__ pt)[64],
    const v16h& qa0, const v16h& qa1, const v16h& ones,
    v8f (&o)[4], float (&mst)[8], float (&lst)[8],
    int l16, int hlf) {
  // ---- scores: 4 x 16-col sub-tiles, contraction d=64 in two chunks ----
  v8f s[4];
#pragma unroll
  for (int nh = 0; nh < 4; ++nh) {
    v8f sa = {};
    v16h kf = load8x2(&Kt[nh * 16 + l16][hlf * 8]);
    sa = __builtin_amdgcn_wmma_f32_16x16x32_f16(false, qa0, false, kf, (short)0, sa, false, false);
    kf = load8x2(&Kt[nh * 16 + l16][32 + hlf * 8]);
    sa = __builtin_amdgcn_wmma_f32_16x16x32_f16(false, qa1, false, kf, (short)0, sa, false, false);
    s[nh] = sa;
  }

  float p[4][8], alpha[8];
#pragma unroll
  for (int r = 0; r < 8; ++r) {
    float sb[4];
#pragma unroll
    for (int nh = 0; nh < 4; ++nh) {
      if constexpr (BIAS_MODE == 0) {
        sb[nh] = s[nh][r] + bc;
      } else {
        const int i_ = qbase + r + 8 * hlf;
        const int d = (kv0 + nh * 16 + l16) - i_;
        float b0 = rp[0];                     // flat selects -> v_cndmask
        b0 = (d >= -1) ? rp[1] : b0;
        b0 = (d >=  0) ? rp[2] : b0;
        b0 = (d >=  1) ? rp[3] : b0;
        b0 = (d >=  2) ? rp[4] : b0;
        sb[nh] = s[nh][r] + b0;
      }
    }
    float rmax = fmaxf(fmaxf(sb[0], sb[1]), fmaxf(sb[2], sb[3]));
#pragma unroll
    for (int msk = 1; msk < 16; msk <<= 1)    // width-16: stays in half-wave
      rmax = fmaxf(rmax, __shfl_xor(rmax, msk, 32));
    const float mn = fmaxf(mst[r], rmax);
    alpha[r] = __expf(mst[r] - mn);
#pragma unroll
    for (int nh = 0; nh < 4; ++nh) p[nh][r] = __expf(sb[nh] - mn);
    mst[r] = mn;
  }

  // ---- reshape P: C-fragment -> LDS -> A-fragments (same wave, DS in-order)
#pragma unroll
  for (int r = 0; r < 8; ++r)
#pragma unroll
    for (int nh = 0; nh < 4; ++nh)
      pt[r + 8 * hlf][nh * 16 + l16] = (f16)p[nh][r];
  const v16h pa0 = load8x2(&pt[l16][hlf * 8]);
  const v16h pa1 = load8x2(&pt[l16][32 + hlf * 8]);

  // ---- row sums via WMMA against ones ----
  v8f srow = {};
  srow = __builtin_amdgcn_wmma_f32_16x16x32_f16(false, pa0, false, ones, (short)0, srow, false, false);
  srow = __builtin_amdgcn_wmma_f32_16x16x32_f16(false, pa1, false, ones, (short)0, srow, false, false);
#pragma unroll
  for (int r = 0; r < 8; ++r) lst[r] = lst[r] * alpha[r] + srow[r];

  // ---- rescale O, then O += P * V (contraction kv=64 in two chunks) ----
#pragma unroll
  for (int c = 0; c < 4; ++c) {
    v8f oc = o[c];
#pragma unroll
    for (int r = 0; r < 8; ++r) oc[r] *= alpha[r];
    const v16h vf0 = load8x2(&Vs[c * 16 + l16][hlf * 8]);
    const v16h vf1 = load8x2(&Vs[c * 16 + l16][32 + hlf * 8]);
    oc   = __builtin_amdgcn_wmma_f32_16x16x32_f16(false, pa0, false, vf0, (short)0, oc, false, false);
    o[c] = __builtin_amdgcn_wmma_f32_16x16x32_f16(false, pa1, false, vf1, (short)0, oc, false, false);
  }
}

__global__ void __launch_bounds__(128)
attn_kernel(const f16* __restrict__ Qp, const f16* __restrict__ Kp,
            const f16* __restrict__ Vt, const float* __restrict__ rel_pos,
            f16* __restrict__ attn) {
  __shared__ f16 Kt[2][64][64];      // 16 KB  K tile  [kv][d], double buffered
  __shared__ f16 Vs[2][64][64];      // 16 KB  V tile  [d][kv], double buffered
  __shared__ f16 p_tile[4][16][64];  //  8 KB  per-wave P bounce

  const int wave = threadIdx.x >> 5;
  const int lane = threadIdx.x & 31;
  const int l16 = lane & 15, hlf = lane >> 4;
  const int qc = blockIdx.x & (SEQ / 64 - 1);      // 64 q-rows per block
  const int bh = blockIdx.x >> 6;
  const int h = bh & (NUM_HEADS - 1), bb = bh >> 3;
  const int qbase = qc * 64 + wave * 16;
  const int qbs = __builtin_amdgcn_readfirstlane(qbase);  // scalar region checks

  float rp[5];
#pragma unroll
  for (int i = 0; i < 5; ++i) rp[i] = rel_pos[h * 5 + i];

  v16h ones;
#pragma unroll
  for (int i = 0; i < 16; ++i) ones[i] = (f16)1.0f;

  const v16h qa0 = load8x2(Qp + (long)(bb * SEQ + qbase + l16) * D_MODEL +
                           h * HEAD_DIM + hlf * 8);
  const v16h qa1 = load8x2(Qp + (long)(bb * SEQ + qbase + l16) * D_MODEL +
                           h * HEAD_DIM + 32 + hlf * 8);

  float mst[8], lst[8];
  v8f o[4];
#pragma unroll
  for (int r = 0; r < 8; ++r) { mst[r] = -INFINITY; lst[r] = 0.f; }
#pragma unroll
  for (int c = 0; c < 4; ++c) { v8f z = {}; o[c] = z; }

  const f16* Kb = Kp + (long)bb * SEQ * D_MODEL + h * HEAD_DIM;
  const f16* Vb = Vt + (long)(bb * NUM_HEADS + h) * HEAD_DIM * SEQ;
  f16 (*pt)[64] = p_tile[wave];

  // cooperative stage: this wave copies 16 rows of K and 16 d-rows of V
  const int srow = wave * 16 + (lane >> 1);
  const int scol = (lane & 1) * 32;
  auto stage = [&](int kv0, int buf) {
#pragma unroll
    for (int j = 0; j < 4; ++j) {
      async_copy16(Kb + (long)(kv0 + srow) * D_MODEL + scol + j * 8,
                   &Kt[buf][srow][scol + j * 8]);
      async_copy16(Vb + (long)srow * SEQ + kv0 + scol + j * 8,
                   &Vs[buf][srow][scol + j * 8]);
    }
  };

  stage(0, 0);
  const int NB = SEQ / 64;
  for (int it = 0; it < NB; ++it) {
    wait_async0();        // own async copies for block `it` complete
    __syncthreads();      // all waves' copies complete; prev compute done
    if (it + 1 < NB) stage((it + 1) * 64, (it + 1) & 1);
    const int kv0 = it * 64;
    const int buf = it & 1;
    // scalar (SGPR) region dispatch: bias uniform away from the diagonal
    if (kv0 >= qbs + 17)
      attn_block64<0>(kv0, qbase, rp, rp[4], Kt[buf], Vs[buf], pt,
                      qa0, qa1, ones, o, mst, lst, l16, hlf);
    else if (kv0 + 63 <= qbs - 2)
      attn_block64<0>(kv0, qbase, rp, rp[0], Kt[buf], Vs[buf], pt,
                      qa0, qa1, ones, o, mst, lst, l16, hlf);
    else
      attn_block64<1>(kv0, qbase, rp, 0.f, Kt[buf], Vs[buf], pt,
                      qa0, qa1, ones, o, mst, lst, l16, hlf);
  }

  // ---- normalize and store f16 attention output [B,S,D] ----
#pragma unroll
  for (int r = 0; r < 8; ++r) {
    const float inv = 1.f / lst[r];
    const int row = qbase + r + 8 * hlf;
#pragma unroll
    for (int c = 0; c < 4; ++c)
      attn[(long)(bb * SEQ + row) * D_MODEL + h * HEAD_DIM + c * 16 + l16] =
          (f16)(o[c][r] * inv);
  }
}

// ---------------------------------------------------------------------------
extern "C" void kernel_launch(void* const* d_in, const int* in_sizes, int n_in,
                              void* d_out, int out_size, void* d_ws, size_t ws_size,
                              hipStream_t stream) {
  const float* q  = (const float*)d_in[0];
  const float* k  = (const float*)d_in[1];
  const float* v  = (const float*)d_in[2];
  const float* Wq = (const float*)d_in[3];
  const float* bq = (const float*)d_in[4];
  const float* Wk = (const float*)d_in[5];
  const float* bk = (const float*)d_in[6];
  const float* Wv = (const float*)d_in[7];
  const float* bv = (const float*)d_in[8];
  const float* Wo = (const float*)d_in[9];
  const float* bo = (const float*)d_in[10];
  const float* rel_pos = (const float*)d_in[11];

  char* ws = (char*)d_ws;
  const size_t WSZ = (size_t)D_MODEL * D_MODEL * sizeof(f16);   // 512 KB
  const size_t MSZ = (size_t)MROWS * D_MODEL * sizeof(f16);     // 8 MB
  f16* Wq16 = (f16*)(ws);
  f16* Wk16 = (f16*)(ws + WSZ);
  f16* Wv16 = (f16*)(ws + 2 * WSZ);
  f16* Wo16 = (f16*)(ws + 3 * WSZ);
  size_t off = 4 * WSZ;
  f16* Qp   = (f16*)(ws + off); off += MSZ;
  f16* Kp   = (f16*)(ws + off); off += MSZ;
  f16* Vt   = (f16*)(ws + off); off += MSZ;
  f16* attn = (f16*)(ws + off); off += MSZ;    // total ~36 MB

  const int nW = D_MODEL * D_MODEL;
  cvt_f32_f16<<<(nW + 255) / 256, 256, 0, stream>>>(Wq, Wq16, nW);
  cvt_f32_f16<<<(nW + 255) / 256, 256, 0, stream>>>(Wk, Wk16, nW);
  cvt_f32_f16<<<(nW + 255) / 256, 256, 0, stream>>>(Wv, Wv16, nW);
  cvt_f32_f16<<<(nW + 255) / 256, 256, 0, stream>>>(Wo, Wo16, nW);

  const int tiles = (MROWS / 16) * (D_MODEL / 64);   // 4096 wave-tiles
  const dim3 gblk(tiles / 8);
  const float qscale = 0.125f;                       // 1/sqrt(64)
  gemm_wmma_kernel<float, 0><<<gblk, 256, 0, stream>>>(q, Wq16, bq, Qp,
      MROWS, D_MODEL, D_MODEL, qscale);
  gemm_wmma_kernel<float, 0><<<gblk, 256, 0, stream>>>(k, Wk16, bk, Kp,
      MROWS, D_MODEL, D_MODEL, 1.0f);
  gemm_wmma_kernel<float, 1><<<gblk, 256, 0, stream>>>(v, Wv16, bv, Vt,
      MROWS, D_MODEL, D_MODEL, 1.0f);

  attn_kernel<<<BATCH * NUM_HEADS * (SEQ / 64), 128, 0, stream>>>(
      Qp, Kp, Vt, rel_pos, attn);

  gemm_wmma_kernel<f16, 2><<<gblk, 256, 0, stream>>>(attn, Wo16, bo, d_out,
      MROWS, D_MODEL, D_MODEL, 1.0f);
}
